// GIN_84645215470228
// MI455X (gfx1250) — matrix operand order repaired and checked by reading, not compiled, not run on hardware
//
#include <hip/hip_runtime.h>
#include <cmath>

// ---------------------------------------------------------------------------
// GIN 2-layer forward for MI455X (gfx1250, wave32, WMMA).
//   agg1 = x + segsum(x[src]->dst) ; h1 = relu(agg1 @ W1 + b1)
//   h2p  = h1 + segsum(h1[src]->dst); out = log_softmax(h2p @ W2 + b2)
// Memory-bound (~2GB traffic @ 23.3TB/s); GEMMs on v_wmma_f32_16x16x32_f16.
// Workspace layout (bytes), total ~245 MB:
//   [0)          W1 packed f16 B-fragments : 16nt*4kb*32lane*16 = 64 KB
//   [65536)      W2 packed f16 B-fragments : 2nt*8kb*32lane*16  = 16 KB
//   [131072)     agg1  fp32 100000x128  (51.2 MB)
//   [OFF_H1)     h1    fp32 100000x256  (102.4 MB)
//   [OFF_H2)     h2pre fp32 100000x256  (102.4 MB)
// ---------------------------------------------------------------------------

typedef __attribute__((ext_vector_type(16))) _Float16 v16h;
typedef __attribute__((ext_vector_type(8)))  _Float16 v8h;
typedef __attribute__((ext_vector_type(8)))  float    v8f;

#define N_NODES 100000
#define N_EDGES 600000
#define F_IN    128
#define F_HID   256
#define F_OUT   24
#define MTILES  6250            // 100000 / 16

static constexpr size_t OFF_W1P  = 0;
static constexpr size_t OFF_W2P  = 65536;
static constexpr size_t OFF_AGG1 = 131072;
static constexpr size_t OFF_H1   = OFF_AGG1 + (size_t)N_NODES * F_IN  * 4;
static constexpr size_t OFF_H2   = OFF_H1   + (size_t)N_NODES * F_HID * 4;

union UA { v16h v; v8h h[2]; _Float16 e[16]; };
union UC { v8f  v; float f[8]; };

// ---------------------------------------------------------------------------
// K1: pack W1 (128x256) and W2 (256x24, N-padded to 32) into f16 WMMA
// B-fragment order: elem i of lane L in fragment (nt,kb) holds
// K = kb*32 + 8*(L>=16) + i + 8*(i>=8), N = nt*16 + (L&15).  One v16h (32B) load/lane.
// ---------------------------------------------------------------------------
__global__ void gin_pack_weights(const float* __restrict__ W1,
                                 const float* __restrict__ W2,
                                 _Float16* __restrict__ W1p,
                                 _Float16* __restrict__ W2p) {
  int idx = blockIdx.x * blockDim.x + threadIdx.x;
  if (idx < 16 * 4 * 32 * 16) {                 // 32768 W1 elements
    int i    = idx & 15;
    int lane = (idx >> 4) & 31;
    int kb   = (idx >> 9) & 3;
    int nt   = idx >> 11;
    int k = kb * 32 + ((lane & 16) ? 8 : 0) + (i < 8 ? i : i + 8);
    int n = nt * 16 + (lane & 15);
    W1p[idx] = (_Float16)W1[k * F_HID + n];
  } else if (idx < 32768 + 2 * 8 * 32 * 16) {   // 8192 W2 elements
    int j    = idx - 32768;
    int i    = j & 15;
    int lane = (j >> 4) & 31;
    int kb   = (j >> 9) & 7;
    int nt   = j >> 12;
    int k = kb * 32 + ((lane & 16) ? 8 : 0) + (i < 8 ? i : i + 8);
    int n = nt * 16 + (lane & 15);
    W2p[j] = (n < F_OUT) ? (_Float16)W2[k * F_OUT + n] : (_Float16)0.0f;
  }
}

// ---------------------------------------------------------------------------
// K2: self term, agg1 = x  (eps = 0).  float4 copy.
// ---------------------------------------------------------------------------
__global__ void gin_init_agg1(const float* __restrict__ x,
                              float* __restrict__ agg1) {
  size_t t = (size_t)blockIdx.x * blockDim.x + threadIdx.x;   // float4 index
  if (t < (size_t)N_NODES * F_IN / 4) {
    const float4 v = ((const float4*)x)[t];
    ((float4*)agg1)[t] = v;
  }
}

// ---------------------------------------------------------------------------
// K3: layer-1 scatter: agg1[dst,:] += x[src,:].  32 lanes per edge,
// float4 gather + 4x global_atomic_add_f32 per lane.
// ---------------------------------------------------------------------------
__global__ void gin_scatter1(const float* __restrict__ x,
                             const long long* __restrict__ eidx,
                             float* __restrict__ agg1) {
  size_t t = (size_t)blockIdx.x * blockDim.x + threadIdx.x;
  size_t e = t >> 5;
  if (e >= N_EDGES) return;
  int c = (int)(t & 31) * 4;
  int src = (int)eidx[e];
  int dst = (int)eidx[N_EDGES + e];
  const float4 v = *(const float4*)(x + (size_t)src * F_IN + c);
  float* p = agg1 + (size_t)dst * F_IN + c;
  unsafeAtomicAdd(p + 0, v.x);
  unsafeAtomicAdd(p + 1, v.y);
  unsafeAtomicAdd(p + 2, v.z);
  unsafeAtomicAdd(p + 3, v.w);
}

// ---------------------------------------------------------------------------
// K4: GEMM1 + bias + ReLU via WMMA.  Block = 256 thr (8 waves) = one M-tile
// of 16 rows x all 256 cols.  A tile (16x128) staged once in LDS as f16;
// each wave owns N-tiles {w, w+8} (2 f32 accumulators), K-loop 4 x k32.
// Writes h1 and the layer-2 self term h2pre (identical values).
// ---------------------------------------------------------------------------
__global__ void gin_gemm1(const float* __restrict__ agg1,
                          const _Float16* __restrict__ W1p,
                          const float* __restrict__ b1,
                          float* __restrict__ h1,
                          float* __restrict__ h2pre) {
  __shared__ _Float16 As[16 * F_IN];            // 4 KB, row-major [16][128]
  const int mtile = blockIdx.x;                 // 0..6249
  const int tid   = threadIdx.x;

  // cooperative A load: 2048 floats / 256 threads = 8 each, f32 -> f16
  {
    int base = tid * 8;
    int row  = base >> 7;
    int col  = base & 127;
    const float* sp = agg1 + ((size_t)(mtile * 16 + row)) * F_IN + col;
    float4 p0 = *(const float4*)sp;
    float4 p1 = *(const float4*)(sp + 4);
    _Float16* dp = &As[row * F_IN + col];
    dp[0] = (_Float16)p0.x; dp[1] = (_Float16)p0.y;
    dp[2] = (_Float16)p0.z; dp[3] = (_Float16)p0.w;
    dp[4] = (_Float16)p1.x; dp[5] = (_Float16)p1.y;
    dp[6] = (_Float16)p1.z; dp[7] = (_Float16)p1.w;
  }
  __syncthreads();

  const int wave = tid >> 5;                    // 0..7 -> ntiles wave, wave+8
  const int lane = tid & 31;
  const int arow  = lane & 15;
  const int khalf = (lane >> 4) * 8;

  v8f acc0 = {}, acc1 = {};
  for (int kb = 0; kb < 4; ++kb) {
    UA ua;
    int abase = arow * F_IN + kb * 32 + khalf;  // 16B aligned
    ua.h[0] = *(const v8h*)(&As[abase]);
    ua.h[1] = *(const v8h*)(&As[abase + 16]);
    const v16h b0 = *(const v16h*)(W1p + ((size_t)((wave      * 4 + kb) * 32 + lane)) * 16);
    const v16h b1v = *(const v16h*)(W1p + ((size_t)(((wave + 8) * 4 + kb) * 32 + lane)) * 16);
    acc0 = __builtin_amdgcn_wmma_f32_16x16x32_f16(false, ua.v, false, b0,
                                                  (short)0, acc0, false, false);
    acc1 = __builtin_amdgcn_wmma_f32_16x16x32_f16(false, ua.v, false, b1v,
                                                  (short)0, acc1, false, false);
  }

  // epilogue: bias + relu, C layout: f[j] -> M = j + 8*(lane>=16), N = lane&15
  UC c0, c1; c0.v = acc0; c1.v = acc1;
  const int col0 = wave * 16 + (lane & 15);
  const int col1 = (wave + 8) * 16 + (lane & 15);
  const float bb0 = b1[col0], bb1 = b1[col1];
  const int rbase = mtile * 16 + ((lane >> 4) << 3);
  for (int j = 0; j < 8; ++j) {
    size_t r = (size_t)(rbase + j) * F_HID;
    float v0 = fmaxf(c0.f[j] + bb0, 0.0f);
    float v1 = fmaxf(c1.f[j] + bb1, 0.0f);
    h1[r + col0] = v0;  h2pre[r + col0] = v0;
    h1[r + col1] = v1;  h2pre[r + col1] = v1;
  }
}

// ---------------------------------------------------------------------------
// K5: layer-2 scatter: h2pre[dst,:] += h1[src,:].  64 lanes per edge.
// ---------------------------------------------------------------------------
__global__ void gin_scatter2(const float* __restrict__ h1,
                             const long long* __restrict__ eidx,
                             float* __restrict__ h2pre) {
  size_t t = (size_t)blockIdx.x * blockDim.x + threadIdx.x;
  size_t e = t >> 6;
  if (e >= N_EDGES) return;
  int c = (int)(t & 63) * 4;
  int src = (int)eidx[e];
  int dst = (int)eidx[N_EDGES + e];
  const float4 v = *(const float4*)(h1 + (size_t)src * F_HID + c);
  float* p = h2pre + (size_t)dst * F_HID + c;
  unsafeAtomicAdd(p + 0, v.x);
  unsafeAtomicAdd(p + 1, v.y);
  unsafeAtomicAdd(p + 2, v.z);
  unsafeAtomicAdd(p + 3, v.w);
}

// ---------------------------------------------------------------------------
// K6: GEMM2 + bias + fused log_softmax.  One wave per M-tile (16 rows),
// 2 accumulators cover N 0..31 (24 valid), K-loop 8 x k32.  Row reductions
// via __shfl_xor within the 16-lane C-matrix groups.
// ---------------------------------------------------------------------------
__global__ void gin_gemm2_lsm(const float* __restrict__ h2pre,
                              const _Float16* __restrict__ W2p,
                              const float* __restrict__ b2,
                              float* __restrict__ out) {
  const int wid   = blockIdx.x * 8 + (threadIdx.x >> 5);
  if (wid >= MTILES) return;
  const int lane  = threadIdx.x & 31;
  const int arow  = lane & 15;
  const int khalf = (lane >> 4) * 8;
  const size_t rowp = (size_t)(wid * 16 + arow) * F_HID;

  v8f acc0 = {}, acc1 = {};
  for (int kb = 0; kb < 8; ++kb) {
    UA ua;
    const float* ap = h2pre + rowp + kb * 32 + khalf;
    float4 f0 = *(const float4*)(ap);
    float4 f1 = *(const float4*)(ap + 4);
    float4 f2 = *(const float4*)(ap + 16);
    float4 f3 = *(const float4*)(ap + 20);
    ua.e[0]  = (_Float16)f0.x; ua.e[1]  = (_Float16)f0.y;
    ua.e[2]  = (_Float16)f0.z; ua.e[3]  = (_Float16)f0.w;
    ua.e[4]  = (_Float16)f1.x; ua.e[5]  = (_Float16)f1.y;
    ua.e[6]  = (_Float16)f1.z; ua.e[7]  = (_Float16)f1.w;
    ua.e[8]  = (_Float16)f2.x; ua.e[9]  = (_Float16)f2.y;
    ua.e[10] = (_Float16)f2.z; ua.e[11] = (_Float16)f2.w;
    ua.e[12] = (_Float16)f3.x; ua.e[13] = (_Float16)f3.y;
    ua.e[14] = (_Float16)f3.z; ua.e[15] = (_Float16)f3.w;
    const v16h b0 = *(const v16h*)(W2p + ((size_t)((0 * 8 + kb) * 32 + lane)) * 16);
    const v16h b1v = *(const v16h*)(W2p + ((size_t)((1 * 8 + kb) * 32 + lane)) * 16);
    acc0 = __builtin_amdgcn_wmma_f32_16x16x32_f16(false, ua.v, false, b0,
                                                  (short)0, acc0, false, false);
    acc1 = __builtin_amdgcn_wmma_f32_16x16x32_f16(false, ua.v, false, b1v,
                                                  (short)0, acc1, false, false);
  }

  UC c0, c1; c0.v = acc0; c1.v = acc1;
  const int nl   = lane & 15;
  const int col0 = nl;                 // always < 24
  const int col1 = 16 + nl;            // valid iff nl < 8
  const bool v1ok = (col1 < F_OUT);
  const float bb0 = b2[col0];
  const float bb1 = v1ok ? b2[col1] : 0.0f;
  const int rbase = wid * 16 + ((lane >> 4) << 3);

  for (int j = 0; j < 8; ++j) {
    float a0 = c0.f[j] + bb0;
    float a1 = v1ok ? (c1.f[j] + bb1) : -INFINITY;
    // row max over the 16-lane group (lanes 0-15 carry rows j, 16-31 rows j+8)
    float m = fmaxf(a0, a1);
    for (int off = 1; off < 16; off <<= 1)
      m = fmaxf(m, __shfl_xor(m, off, 16));
    float s = __expf(a0 - m) + (v1ok ? __expf(a1 - m) : 0.0f);
    for (int off = 1; off < 16; off <<= 1)
      s += __shfl_xor(s, off, 16);
    float lse = m + __logf(s);
    size_t r = (size_t)(rbase + j) * F_OUT;
    out[r + col0] = a0 - lse;
    if (v1ok) out[r + col1] = a1 - lse;
  }
}

// ---------------------------------------------------------------------------
extern "C" void kernel_launch(void* const* d_in, const int* in_sizes, int n_in,
                              void* d_out, int out_size, void* d_ws, size_t ws_size,
                              hipStream_t stream) {
  const float*     x    = (const float*)d_in[0];
  const long long* eidx = (const long long*)d_in[1];   // int64 (2, N_EDGES)
  const float*     W1   = (const float*)d_in[2];
  const float*     b1   = (const float*)d_in[3];
  const float*     W2   = (const float*)d_in[4];
  const float*     b2   = (const float*)d_in[5];
  float*           out  = (float*)d_out;

  char* ws = (char*)d_ws;
  _Float16* W1p   = (_Float16*)(ws + OFF_W1P);
  _Float16* W2p   = (_Float16*)(ws + OFF_W2P);
  float*    agg1  = (float*)(ws + OFF_AGG1);
  float*    h1    = (float*)(ws + OFF_H1);
  float*    h2pre = (float*)(ws + OFF_H2);

  // K1: weight packing (40960 elements)
  gin_pack_weights<<<160, 256, 0, stream>>>(W1, W2, W1p, W2p);
  // K2: agg1 = x  (3.2M float4)
  gin_init_agg1<<<12500, 256, 0, stream>>>(x, agg1);
  // K3: scatter layer 1 (600000 edges x 32 lanes)
  gin_scatter1<<<75000, 256, 0, stream>>>(x, eidx, agg1);
  // K4: WMMA GEMM1 + bias + relu (one block per 16-row M-tile)
  gin_gemm1<<<MTILES, 256, 0, stream>>>(agg1, W1p, b1, h1, h2pre);
  // K5: scatter layer 2 (600000 edges x 64 lanes)
  gin_scatter2<<<150000, 256, 0, stream>>>(h1, eidx, h2pre);
  // K6: WMMA GEMM2 + bias + log_softmax (one wave per M-tile, 8 waves/block)
  gin_gemm2_lsm<<<(MTILES + 7) / 8, 256, 0, stream>>>(h2pre, W2p, b2, out);
}
